// EquivariantBlock_21560735826062
// MI455X (gfx1250) — matrix operand order repaired and checked by reading, not compiled, not run on hardware
//
#include <hip/hip_runtime.h>
#include <hip/hip_bf16.h>

// ---------------------------------------------------------------------------
// EGNN EquivariantBlock for MI455X (gfx1250, wave32, WMMA bf16 K=32).
// h (50000x128 fp32 = 25.6MB) is L2-resident -> edge gathers hit L2; the
// problem is WMMA-throughput bound, so all GEMMs run as v_wmma_f32_16x16x32_bf16
// with f32 accumulation. One wave owns a 64-row tile (4 x 16 M-tiles) so each
// B fragment load pair feeds four WMMAs (load-path is the binding resource on
// the shared CU memory pipe). 1 wave per block, LDS staging ~37KB.
// ---------------------------------------------------------------------------

typedef __bf16 v16bf __attribute__((ext_vector_type(16)));
typedef float  v8f   __attribute__((ext_vector_type(8)));

union Frag {
    v16bf    v;
    unsigned u[8];
    uint4    q[2];
};

static constexpr int HD    = 128;  // hidden
static constexpr int S1e   = 292;  // LDS row stride (shorts) edge staging (288 pad + skew)
static constexpr int S2n   = 260;  // LDS row stride (shorts) node staging (256 + skew)
static constexpr int KC_E1 = 9;    // ceil(258/32) -> K padded to 288
static constexpr int KC_N1 = 8;    // 256/32
static constexpr int KC_H  = 4;    // 128/32
static constexpr int MT    = 4;    // M-tiles (16 rows each) per wave

__device__ __forceinline__ unsigned short f2bf(float f) {
    union { __bf16 b; unsigned short u; } v;
    v.b = (__bf16)f;                 // native v_cvt bf16 on gfx1250
    return v.u;
}

__device__ __forceinline__ float sigf(float v) {
    return 1.0f / (1.0f + __expf(-v));
}

// ---------------------------------------------------------------------------
// A-fragment: 16-bit A 16x32 layout (ISA 7.12.2): lane m=lane&15, halfk=lane>>4,
// VGPR r holds K pair: K = (r>>2)*16 + (r&3)*2 + halfk*8   (+ kc*32 via rowp).
// ---------------------------------------------------------------------------
__device__ __forceinline__ void load_a(Frag& a, const unsigned short* rowp, int hk) {
#pragma unroll
    for (int r = 0; r < 8; ++r) {
        int k = ((r >> 2) << 4) + ((r & 3) << 1) + (hk << 3);
        a.u[r] = *(const unsigned*)(rowp + k);
    }
}

// B fragments prepacked in global memory as [nt][kc][lane][8 dwords]
__device__ __forceinline__ void load_b(Frag& b, const unsigned* __restrict__ wp,
                                       int nt, int kc, int kcnt, int lane) {
    const uint4* bp = (const uint4*)(wp + ((((nt * kcnt) + kc) * 32 + lane) << 3));
    b.q[0] = bp[0];
    b.q[1] = bp[1];
}

// acc[mt][nt] += Atile(MT x 16xK) * Wtile(Kx16); each B fragment feeds MT WMMAs.
template <int KCNT>
__device__ __forceinline__ void gemm_tileM(const unsigned short* st, int stride,
                                           const unsigned* __restrict__ wp,
                                           int lane, v8f (*acc)[8]) {
    const int m  = lane & 15;
    const int hk = lane >> 4;
    for (int kc = 0; kc < KCNT; ++kc) {
        Frag a[MT];
#pragma unroll
        for (int mt = 0; mt < MT; ++mt)
            load_a(a[mt], st + (m + mt * 16) * stride + kc * 32, hk);
#pragma unroll
        for (int nt = 0; nt < 8; ++nt) {
            Frag b;
            load_b(b, wp, nt, kc, KCNT, lane);
#pragma unroll
            for (int mt = 0; mt < MT; ++mt)
                acc[mt][nt] = __builtin_amdgcn_wmma_f32_16x16x32_bf16(
                    false, a[mt].v, false, b.v, (short)0, acc[mt][nt], false, false);
        }
    }
}

// bias + SiLU, write 16x128 bf16 tile into LDS at row offset rowoff
// (C/D layout: lane n=lane&15, VGPR r -> row (lane>>4)*8 + r)
__device__ __forceinline__ void silu_store(const v8f* acc, const float* __restrict__ bias,
                                           unsigned short* st, int stride, int lane,
                                           int rowoff) {
    const int n  = lane & 15;
    const int mh = (lane >> 4) * 8 + rowoff;
#pragma unroll
    for (int nt = 0; nt < 8; ++nt) {
        float bb = bias[nt * 16 + n];
#pragma unroll
        for (int r = 0; r < 8; ++r) {
            float v = acc[nt][r] + bb;
            v = v * sigf(v);
            st[(mh + r) * stride + nt * 16 + n] = f2bf(v);
        }
    }
}

// ---------------------------------------------------------------------------
// Weight repack: fp32 [128][kin] row-major -> bf16 B-fragment order
// [nt][kc][lane][r], lane: n=lane&15 (column), halfk=lane>>4,
// dword r = { W[n][k0], W[n][k0+1] }, k0 = kc*32 + (r>>2)*16 + (r&3)*2 + halfk*8.
// ---------------------------------------------------------------------------
__global__ void repack_kernel(const float* __restrict__ w, unsigned* __restrict__ dst,
                              int kin, int kcnt) {
    int idx   = blockIdx.x * 256 + threadIdx.x;
    int total = 8 * kcnt * 32 * 8;
    if (idx >= total) return;
    int r = idx & 7, lane = (idx >> 3) & 31, rest = idx >> 8;
    int kc = rest % kcnt, nt = rest / kcnt;
    int n = lane & 15, hk = lane >> 4;
    int k0 = kc * 32 + ((r >> 2) << 4) + ((r & 3) << 1) + (hk << 3);
    int o  = nt * 16 + n;
    unsigned lo = (k0 < kin)     ? (unsigned)f2bf(w[o * kin + k0])     : 0u;
    unsigned hi = (k0 + 1 < kin) ? (unsigned)f2bf(w[o * kin + k0 + 1]) : 0u;
    dst[idx] = lo | (hi << 16);
}

__global__ void init_h_kernel(const float* __restrict__ h, float* __restrict__ hcur,
                              unsigned short* __restrict__ hbf, int nElem) {
    int i = blockIdx.x * 256 + threadIdx.x;
    if (i >= nElem) return;
    float v = h[i];
    hcur[i] = v;
    hbf[i]  = f2bf(v);
}

__global__ void radial_kernel(const float* __restrict__ x, const int* __restrict__ row,
                              const int* __restrict__ col, float* __restrict__ radial, int E) {
    int e = blockIdx.x * 256 + threadIdx.x;
    if (e >= E) return;
    int r = row[e], c = col[e];
    float dx = x[r * 3 + 0] - x[c * 3 + 0];
    float dy = x[r * 3 + 1] - x[c * 3 + 1];
    float dz = x[r * 3 + 2] - x[c * 3 + 2];
    radial[e] = dx * dx + dy * dy + dz * dz;
}

__global__ void finish_x_kernel(const float* __restrict__ x, const float* __restrict__ xacc,
                                float* __restrict__ xout, int nElem) {
    int i = blockIdx.x * 256 + threadIdx.x;
    if (i >= nElem) return;
    xout[i] = x[i] + xacc[i] * 0.01f;   // / NORM_FACTOR
}

// ---------------------------------------------------------------------------
// Edge MLP. MODE 0: GCL edge model (attention gate + scatter into agg).
//           MODE 1: Equivariant update (dot w3, scatter coord_diff*s into xacc).
// One wave (= one block, 32 threads) owns 64 edges (4 M-tiles).
// ---------------------------------------------------------------------------
template <int MODE>
__global__ __launch_bounds__(32) void edge_mlp_kernel(
    const unsigned short* __restrict__ hbf, const int* __restrict__ row,
    const int* __restrict__ col, const float* __restrict__ radial,
    const float* __restrict__ eattr, const float* __restrict__ x,
    const unsigned* __restrict__ w1p, const unsigned* __restrict__ w2p,
    const float* __restrict__ b1, const float* __restrict__ b2,
    const float* __restrict__ awv, const float* __restrict__ abv,
    const float* __restrict__ w3, float* __restrict__ agg,
    float* __restrict__ xacc, int nEdges) {
    __shared__ __align__(16) unsigned short st[64 * S1e];   // ~37.4 KB
    __shared__ int rl[64], cl[64];

    const int lane = threadIdx.x & 31;
    const int e0   = blockIdx.x * 64;
    if (e0 >= nEdges) return;
    const int n  = lane & 15;
    const int mh = (lane >> 4) * 8;

    // ---- stage edge metadata + extra features + zero pad (K 258..291)
#pragma unroll
    for (int jj = 0; jj < 2; ++jj) {
        int j = lane + jj * 32;
        int e = e0 + j;
        if (e >= nEdges) e = nEdges - 1;        // clamp (scatter is guarded below)
        rl[j] = row[e];
        cl[j] = col[e];
        unsigned short* pp = st + j * S1e;
        pp[256] = f2bf(radial[e]);
        pp[257] = f2bf(eattr[e]);
        unsigned* pz = (unsigned*)(pp + 258);
#pragma unroll
        for (int i = 0; i < 17; ++i) pz[i] = 0u;
    }
    // ---- gather h[row], h[col] (bf16) into LDS rows; same-wave DS is in-order
#pragma unroll 4
    for (int j = 0; j < 64; ++j) {
        int r = rl[j], c = cl[j];
        uint2 hr = *(const uint2*)(hbf + (size_t)r * HD + lane * 4);
        uint2 hc = *(const uint2*)(hbf + (size_t)c * HD + lane * 4);
        *(uint2*)(st + j * S1e + lane * 4)      = hr;
        *(uint2*)(st + j * S1e + HD + lane * 4) = hc;
    }

    // ---- GEMM1: [64 x 288] x [288 x 128]  (288 WMMAs, B reused x4)
    v8f acc[MT][8];
#pragma unroll
    for (int mt = 0; mt < MT; ++mt)
#pragma unroll
        for (int t = 0; t < 8; ++t) acc[mt][t] = (v8f){0, 0, 0, 0, 0, 0, 0, 0};
    gemm_tileM<KC_E1>(st, S1e, w1p, lane, acc);
#pragma unroll
    for (int mt = 0; mt < MT; ++mt)
        silu_store(acc[mt], b1, st, S1e, lane, mt * 16);   // mij1 overwrites e_in rows

    // ---- GEMM2: [64 x 128] x [128 x 128]  (128 WMMAs)
    v8f acc2[MT][8];
#pragma unroll
    for (int mt = 0; mt < MT; ++mt)
#pragma unroll
        for (int t = 0; t < 8; ++t) acc2[mt][t] = (v8f){0, 0, 0, 0, 0, 0, 0, 0};
    gemm_tileM<KC_H>(st, S1e, w2p, lane, acc2);

    // ---- epilogue 2 per M-tile: bias + SiLU, dot with aw (MODE 0) or eq_w3 (MODE 1)
#pragma unroll
    for (int mt = 0; mt < MT; ++mt) {
        float mv[8][8];
        float p[8];
#pragma unroll
        for (int r = 0; r < 8; ++r) p[r] = 0.0f;
#pragma unroll
        for (int nt = 0; nt < 8; ++nt) {
            float bb = b2[nt * 16 + n];
            float wn = (MODE == 0) ? awv[nt * 16 + n] : w3[nt * 16 + n];
#pragma unroll
            for (int r = 0; r < 8; ++r) {
                float v = acc2[mt][nt][r] + bb;
                v = v * sigf(v);
                mv[nt][r] = v;
                p[r] += v * wn;
            }
        }
        // reduce over 16 lanes of each half (rows mh..mh+7 of this M-tile)
#pragma unroll
        for (int r = 0; r < 8; ++r) {
            float v = p[r];
            v += __shfl_xor(v, 1, 32);
            v += __shfl_xor(v, 2, 32);
            v += __shfl_xor(v, 4, 32);
            v += __shfl_xor(v, 8, 32);
            p[r] = v;
        }

        if (MODE == 0) {
            float ab0 = abv[0];
#pragma unroll
            for (int r = 0; r < 8; ++r) p[r] = sigf(p[r] + ab0);   // attention gate
#pragma unroll
            for (int r = 0; r < 8; ++r) {
                int mm = mt * 16 + mh + r;
                if (e0 + mm < nEdges) {
                    int rowm   = rl[mm];
                    float* dst = agg + (size_t)rowm * HD + n;
#pragma unroll
                    for (int nt = 0; nt < 8; ++nt)
                        atomicAdd(dst + nt * 16, mv[nt][r] * p[r]);    // segment_sum
                }
            }
        } else {
            if (n == 0) {   // lanes 0 / 16 each own 8 edges of this M-tile
#pragma unroll
                for (int r = 0; r < 8; ++r) {
                    int mm = mt * 16 + mh + r;
                    if (e0 + mm < nEdges) {
                        int rr = rl[mm], cc = cl[mm];
                        float dx  = x[rr * 3 + 0] - x[cc * 3 + 0];
                        float dy  = x[rr * 3 + 1] - x[cc * 3 + 1];
                        float dz  = x[rr * 3 + 2] - x[cc * 3 + 2];
                        float rad = dx * dx + dy * dy + dz * dz;
                        float s   = p[r] / (sqrtf(rad + 1e-8f) + 1.0f);  // NORM_CONSTANT=1
                        atomicAdd(&xacc[rr * 3 + 0], dx * s);
                        atomicAdd(&xacc[rr * 3 + 1], dy * s);
                        atomicAdd(&xacc[rr * 3 + 2], dz * s);
                    }
                }
            }
        }
    }
}

// ---------------------------------------------------------------------------
// Node MLP: h += silu([h | agg/100] @ nw1^T + nb1) @ nw2^T + nb2
// One wave (= one block) owns 64 nodes (4 M-tiles); rewrites hcur + hbf.
// ---------------------------------------------------------------------------
__global__ __launch_bounds__(32) void node_mlp_kernel(
    float* __restrict__ hcur, unsigned short* __restrict__ hbf,
    const float* __restrict__ agg, const unsigned* __restrict__ w1p,
    const unsigned* __restrict__ w2p, const float* __restrict__ b1,
    const float* __restrict__ b2, int nNodes) {
    __shared__ __align__(16) unsigned short st[64 * S2n];   // ~33.3 KB

    const int lane = threadIdx.x & 31;
    const int t0   = blockIdx.x * 64;
    if (t0 >= nNodes) return;
    const int n  = lane & 15;
    const int mh = (lane >> 4) * 8;

    // ---- stage A = [h_bf | bf16(agg * 0.01)] (clamped; stores guarded below)
#pragma unroll 4
    for (int j = 0; j < 64; ++j) {
        int node = t0 + j;
        if (node >= nNodes) node = nNodes - 1;
        uint2 hr = *(const uint2*)(hbf + (size_t)node * HD + lane * 4);
        *(uint2*)(st + j * S2n + lane * 4) = hr;
        float4 av = *(const float4*)(agg + (size_t)node * HD + lane * 4);
        uint2 pk;
        pk.x = (unsigned)f2bf(av.x * 0.01f) | ((unsigned)f2bf(av.y * 0.01f) << 16);
        pk.y = (unsigned)f2bf(av.z * 0.01f) | ((unsigned)f2bf(av.w * 0.01f) << 16);
        *(uint2*)(st + j * S2n + HD + lane * 4) = pk;
    }

    // ---- GEMM1: [64 x 256] x [256 x 128]  (256 WMMAs)
    v8f acc[MT][8];
#pragma unroll
    for (int mt = 0; mt < MT; ++mt)
#pragma unroll
        for (int t = 0; t < 8; ++t) acc[mt][t] = (v8f){0, 0, 0, 0, 0, 0, 0, 0};
    gemm_tileM<KC_N1>(st, S2n, w1p, lane, acc);
#pragma unroll
    for (int mt = 0; mt < MT; ++mt)
        silu_store(acc[mt], b1, st, S2n, lane, mt * 16);

    // ---- GEMM2: [64 x 128] x [128 x 128]  (128 WMMAs)
    v8f acc2[MT][8];
#pragma unroll
    for (int mt = 0; mt < MT; ++mt)
#pragma unroll
        for (int t = 0; t < 8; ++t) acc2[mt][t] = (v8f){0, 0, 0, 0, 0, 0, 0, 0};
    gemm_tileM<KC_H>(st, S2n, w2p, lane, acc2);

    // ---- residual add, write fp32 + bf16 mirror
#pragma unroll
    for (int mt = 0; mt < MT; ++mt) {
#pragma unroll
        for (int nt = 0; nt < 8; ++nt) {
            float bb = b2[nt * 16 + n];
#pragma unroll
            for (int r = 0; r < 8; ++r) {
                int node = t0 + mt * 16 + mh + r;
                if (node < nNodes) {
                    size_t ci = (size_t)node * HD + nt * 16 + n;
                    float v   = acc2[mt][nt][r] + bb + hcur[ci];
                    hcur[ci]  = v;
                    hbf[ci]   = f2bf(v);
                }
            }
        }
    }
}

// ---------------------------------------------------------------------------
extern "C" void kernel_launch(void* const* d_in, const int* in_sizes, int n_in,
                              void* d_out, int out_size, void* d_ws, size_t ws_size,
                              hipStream_t stream) {
    const float* h     = (const float*)d_in[0];
    const float* x     = (const float*)d_in[1];
    const float* eattr = (const float*)d_in[2];
    const int*   eidx  = (const int*)d_in[3];

    const int E = in_sizes[2];        // edge_attr is [E,1]
    const int N = in_sizes[0] / HD;   // h is [N,128]
    const int* row = eidx;
    const int* col = eidx + E;

    const float* W[3][4]  = { { (const float*)d_in[4],  (const float*)d_in[6],
                                (const float*)d_in[10], (const float*)d_in[12] },
                              { (const float*)d_in[14], (const float*)d_in[16],
                                (const float*)d_in[20], (const float*)d_in[22] },
                              { (const float*)d_in[24], (const float*)d_in[26], nullptr, nullptr } };
    const float* B1[3] = { (const float*)d_in[5],  (const float*)d_in[15], (const float*)d_in[25] };
    const float* B2[3] = { (const float*)d_in[7],  (const float*)d_in[17], (const float*)d_in[27] };
    const float* AW[2] = { (const float*)d_in[8],  (const float*)d_in[18] };
    const float* AB[2] = { (const float*)d_in[9],  (const float*)d_in[19] };
    const float* NB1[2] = { (const float*)d_in[11], (const float*)d_in[21] };
    const float* NB2[2] = { (const float*)d_in[13], (const float*)d_in[23] };
    const float* EQW3 = (const float*)d_in[28];

    // final h lives directly in d_out; x tail follows
    float* hcur = (float*)d_out;
    float* xout = (float*)d_out + (size_t)N * HD;

    // ---- workspace carve-up (16B aligned)
    char* ws = (char*)d_ws;
    size_t off = 0;
    auto take = [&](size_t bytes) { char* p = ws + off; off = (off + bytes + 15) & ~(size_t)15; return p; };
    unsigned short* hbf = (unsigned short*)take((size_t)N * HD * 2);
    float* agg    = (float*)take((size_t)N * HD * 4);
    float* xacc   = (float*)take((size_t)N * 3 * 4);
    float* radial = (float*)take((size_t)E * 4);
    // packed weights: kcnt fragments of 2048 dwords each
    auto packsz = [](int kcnt) { return (size_t)2048 * kcnt * 4; };
    unsigned* P[3][4];
    const int KIN[4] = { 258, 128, 256, 128 };
    const int KCN[4] = { KC_E1, KC_H, KC_N1, KC_H };
    for (int l = 0; l < 3; ++l)
        for (int s = 0; s < 4; ++s)
            P[l][s] = W[l][s] ? (unsigned*)take(packsz(KCN[s])) : nullptr;
    (void)ws_size;

    // ---- repack all weight matrices into WMMA B-fragment order
    for (int l = 0; l < 3; ++l)
        for (int s = 0; s < 4; ++s)
            if (W[l][s]) {
                int total = 2048 * KCN[s];
                repack_kernel<<<(total + 255) / 256, 256, 0, stream>>>(W[l][s], P[l][s], KIN[s], KCN[s]);
            }

    init_h_kernel<<<((N * HD) + 255) / 256, 256, 0, stream>>>(h, hcur, hbf, N * HD);
    radial_kernel<<<(E + 255) / 256, 256, 0, stream>>>(x, row, col, radial, E);

    const int eBlocks = (E + 63) / 64;   // 64 edges per wave, 1 wave per block
    const int nBlocks = (N + 63) / 64;   // 64 nodes per wave

    for (int l = 0; l < 2; ++l) {
        hipMemsetAsync(agg, 0, (size_t)N * HD * 4, stream);
        edge_mlp_kernel<0><<<eBlocks, 32, 0, stream>>>(
            hbf, row, col, radial, eattr, x, P[l][0], P[l][1], B1[l], B2[l],
            AW[l], AB[l], nullptr, agg, nullptr, E);
        node_mlp_kernel<<<nBlocks, 32, 0, stream>>>(
            hcur, hbf, agg, P[l][2], P[l][3], NB1[l], NB2[l], N);
    }

    hipMemsetAsync(xacc, 0, (size_t)N * 3 * 4, stream);
    edge_mlp_kernel<1><<<eBlocks, 32, 0, stream>>>(
        hbf, row, col, radial, eattr, x, P[2][0], P[2][1], B1[2], B2[2],
        nullptr, nullptr, EQW3, nullptr, xacc, E);

    finish_x_kernel<<<((N * 3) + 255) / 256, 256, 0, stream>>>(x, xacc, xout, N * 3);
}
